// Attention_80908593922521
// MI455X (gfx1250) — matrix operand order repaired and checked by reading, not compiled, not run on hardware
//
#include <hip/hip_runtime.h>
#include <hip/hip_bf16.h>

// ---------------------------------------------------------------------------
// Problem constants (from the reference): B=2, S=4096, D=128, DI=1024, W=128
// h = 8 heads, dh = 128, n = 32 windows, 3W = 384 keys per window.
// ---------------------------------------------------------------------------
#define BB   2
#define SS   4096
#define DD   128
#define DI_  1024
#define WW   128
#define NH   8
#define DH   128
#define NW   32          // SS / WW
#define MROWS (BB * SS)  // 8192

typedef __attribute__((ext_vector_type(16))) _Float16 v16h;
typedef __attribute__((ext_vector_type(8)))  float    v8f;

union F16Frag {
    v16h     v;
    _Float16 h[16];
    uint4    u[2];   // two 16-byte chunks (halves 0-7, 8-15)
};

__device__ __forceinline__ v8f wmma_f16(v16h a, v16h b, v8f c) {
    // D = A(16x32 f16) * B(32x16 f16) + C(16x16 f32)
    return __builtin_amdgcn_wmma_f32_16x16x32_f16(
        /*neg_a=*/false, a, /*neg_b=*/false, b,
        /*c_mod=*/(short)0, c, /*reuse_a=*/false, /*reuse_b=*/false);
}

// Async global -> LDS copy of 16 bytes (CDNA5 ASYNCcnt path).
// ldsOff = low 32 bits of a generic pointer into __shared__ (== DS byte offset).
__device__ __forceinline__ void async_copy_b128(unsigned ldsOff, const void* gptr) {
    asm volatile("global_load_async_to_lds_b128 %0, %1, off"
                 :: "v"(ldsOff), "v"(gptr) : "memory");
}
__device__ __forceinline__ void async_wait0() {
    asm volatile("s_wait_asynccnt 0x0" ::: "memory");
}

// ---------------------------------------------------------------------------
// f32 -> f16 conversion (grid-stride)
// ---------------------------------------------------------------------------
__global__ void cvt_f32_to_f16(const float* __restrict__ src,
                               _Float16* __restrict__ dst, long long n) {
    long long i = (long long)blockIdx.x * blockDim.x + threadIdx.x;
    long long stride = (long long)gridDim.x * blockDim.x;
    for (; i < n; i += stride) dst[i] = (_Float16)src[i];
}

// ---------------------------------------------------------------------------
// Generic WMMA GEMM:  C[M,N] = A[M,K] @ Bw[N,K]^T   (A,Bw f16 row-major)
// mode 0: C16 = acc
// mode 1: C16 = (acc + bias[n]) * scale          (q projection, scale=1/sqrt(dh))
// mode 2: C16 = sigmoid(acc + bias[n])           (gate)
// mode 3: C32 = acc                              (final output, f32)
// One wave computes a 16x64 tile; K-loop is software pipelined (2 buffers)
// so fragment loads for step k+1 overlap the WMMAs of step k.
// N multiple of 64; K multiple of 64 (K/32 even: 128 and 1024 here).
// ---------------------------------------------------------------------------
__global__ __launch_bounds__(256) void gemm_wmma_kernel(
    const _Float16* __restrict__ A, const _Float16* __restrict__ Bw,
    const float* __restrict__ bias, void* __restrict__ Cout,
    int M, int N, int K, float scale, int mode)
{
    const int lane = threadIdx.x & 31;
    const int wave = threadIdx.x >> 5;
    const int c    = lane & 15;   // column / row-within-tile
    const int hi   = lane >> 4;   // half-wave select

    const int  Nt     = N >> 6;                    // 64-wide column groups
    const long long nTiles = (long long)(M >> 4) * Nt;
    long long tile = (long long)blockIdx.x * 8 + wave;
    if (tile >= nTiles) return;
    const int mt = (int)(tile / Nt);
    const int nt = (int)(tile % Nt);

    const _Float16* arow = A + ((long long)mt * 16 + c) * K;
    const _Float16* bptr[4];
#pragma unroll
    for (int j = 0; j < 4; ++j)
        bptr[j] = Bw + (long long)(nt * 64 + j * 16 + c) * K + hi * 16;

    v8f acc[4];
#pragma unroll
    for (int j = 0; j < 4; ++j)
#pragma unroll
        for (int r = 0; r < 8; ++r) acc[j][r] = 0.0f;

    const int nK = K >> 5;
    F16Frag a0, a1, b0[4], b1[4];

#define LOAD_A(f, ks)  { const int d0 = (ks) * 32 + hi * 16;                  \
                         (f).u[0] = *(const uint4*)(arow + d0);               \
                         (f).u[1] = *(const uint4*)(arow + d0 + 8); }
#define LOAD_B(f, ks)  { _Pragma("unroll")                                    \
                         for (int j = 0; j < 4; ++j) {                        \
                             const _Float16* p = bptr[j] + (ks) * 32;         \
                             (f)[j].u[0] = *(const uint4*)(p);                \
                             (f)[j].u[1] = *(const uint4*)(p + 8); } }

    LOAD_A(a0, 0) LOAD_B(b0, 0)
    for (int ks = 0; ks < nK; ks += 2) {
        if (ks + 1 < nK) { LOAD_A(a1, ks + 1) LOAD_B(b1, ks + 1) }
#pragma unroll
        for (int j = 0; j < 4; ++j) acc[j] = wmma_f16(a0.v, b0[j].v, acc[j]);
        if (ks + 2 < nK) { LOAD_A(a0, ks + 2) LOAD_B(b0, ks + 2) }
        if (ks + 1 < nK) {
#pragma unroll
            for (int j = 0; j < 4; ++j) acc[j] = wmma_f16(a1.v, b1[j].v, acc[j]);
        }
    }
#undef LOAD_A
#undef LOAD_B

#pragma unroll
    for (int j = 0; j < 4; ++j) {
        const int ncol = nt * 64 + j * 16 + c;
        const float bval = bias ? bias[ncol] : 0.0f;
#pragma unroll
        for (int r = 0; r < 8; ++r) {
            const long long m = (long long)mt * 16 + r + 8 * hi;
            float v = acc[j][r];
            if (mode == 1)      v = (v + bval) * scale;
            else if (mode == 2) v = 1.0f / (1.0f + __expf(-(v + bval)));
            if (mode == 3) ((float*)Cout)[m * N + ncol] = v;
            else           ((_Float16*)Cout)[m * N + ncol] = (_Float16)v;
        }
    }
}

// ---------------------------------------------------------------------------
// Windowed attention with online softmax.
// 1 block = (b, head, window n). 8 waves x 16 query rows = 128 rows (W).
// Keys: windows {n-1, n, n+1} ∩ [0,NW), processed in 32-key chunks.
// Per chunk, the K tile (32x128, shared by all 8 waves) is staged into LDS
// via global_load_async_to_lds_b128 (ASYNCcnt), and the V tile is staged
// TRANSPOSED (Vt[d][j]) so both B-fragments are contiguous 32B LDS reads.
// q16 is pre-scaled by dh^-0.5.  Output = softmax(qk + bias, mask) @ v * gate.
// ---------------------------------------------------------------------------
__global__ __launch_bounds__(256) void attn_window_kernel(
    const _Float16* __restrict__ q16, const _Float16* __restrict__ k16,
    const _Float16* __restrict__ v16, const _Float16* __restrict__ g16,
    const float* __restrict__ attn_bias, const unsigned char* __restrict__ mask,
    _Float16* __restrict__ attout)
{
    __shared__ _Float16 k_lds[32][128];     // K tile  [key j][d]      8 KB
    __shared__ _Float16 vt_lds[128][32];    // V tile  [d][key j]      8 KB (transposed)
    __shared__ _Float16 p_lds[8][16][32];   // per-wave probability    8 KB

    const int tid  = threadIdx.x;
    const int lane = tid & 31;
    const int wave = tid >> 5;
    const int c    = lane & 15;
    const int hi   = lane >> 4;

    const int n    = blockIdx.x % NW;
    const int head = (blockIdx.x / NW) % NH;
    const int b    = blockIdx.x / (NW * NH);

    const long long browbase = (long long)b * SS;     // row base into [B*S, DI]
    const int q0 = n * WW + wave * 16;                // first query row (seq pos)

    // ---- load q A-fragments (16 rows x 128 d, 4 K-steps of 32) ----
    F16Frag qA[4];
    {
        const _Float16* qrow = q16 + (browbase + q0 + c) * DI_ + head * DH;
#pragma unroll
        for (int ks = 0; ks < 4; ++ks) {
            const int d0 = ks * 32 + hi * 16;
            qA[ks].u[0] = *(const uint4*)(qrow + d0);
            qA[ks].u[1] = *(const uint4*)(qrow + d0 + 8);
        }
    }

    float m_run[8], l_run[8];
    v8f   acc[8];                                     // 8 d-tiles of 16
#pragma unroll
    for (int r = 0; r < 8; ++r) { m_run[r] = -3.0e38f; l_run[r] = 0.0f; }
#pragma unroll
    for (int dt = 0; dt < 8; ++dt)
#pragma unroll
        for (int r = 0; r < 8; ++r) acc[dt][r] = 0.0f;

    for (int w = -1; w <= 1; ++w) {
        const int wi = n + w;
        if (wi < 0 || wi >= NW) continue;             // padded window -> fully masked
        const int kwbase = wi * WW;                   // key window base (seq pos)

        for (int jt = 0; jt < 4; ++jt) {              // 32 keys per chunk
            const int jb = jt * 32;

            // ===== cooperative staging of K (async -> LDS) and Vt =====
            __syncthreads();   // WAR: previous chunk fully consumed
            {
                // K: 32 rows x 256B = 512 x 16B chunks; 2 per thread
#pragma unroll
                for (int i = 0; i < 2; ++i) {
                    const int chunk = i * 256 + tid;       // 0..511
                    const int j    = chunk >> 4;           // key row 0..31
                    const int doff = (chunk & 15) * 8;     // half offset in row
                    const _Float16* gk =
                        k16 + (browbase + kwbase + jb + j) * DI_ + head * DH + doff;
                    const unsigned ldsOff =
                        (unsigned)(unsigned long long)&k_lds[j][doff];
                    async_copy_b128(ldsOff, gk);
                }
                // V: coalesced 32B global read per thread, transposed 2B DS stores
                const int j    = tid >> 3;                 // key row 0..31
                const int d0   = (tid & 7) * 16;           // 16 halves of d
                const _Float16* gv =
                    v16 + (browbase + kwbase + jb + j) * DI_ + head * DH + d0;
                F16Frag vv;
                vv.u[0] = *(const uint4*)(gv);
                vv.u[1] = *(const uint4*)(gv + 8);
#pragma unroll
                for (int e = 0; e < 16; ++e) vt_lds[d0 + e][j] = vv.h[e];
            }
            // prefetch next chunk's attn_bias lines (lane c covers query row q0+c)
            if (jt < 3)
                __builtin_prefetch(
                    attn_bias + ((long long)b * SS + q0 + c) * SS + kwbase + jb + 32, 0, 0);
            async_wait0();     // my wave's async K chunks landed in LDS
            __syncthreads();   // everyone's K + Vt staged

            // ===== scores for two 16-key subtiles =====
            v8f   sfr[2];
            float tmax[8];
#pragma unroll
            for (int r = 0; r < 8; ++r) tmax[r] = -3.0e38f;

#pragma unroll
            for (int js = 0; js < 2; ++js) {
                const int jl   = js * 16 + c;              // local key for my column
                const int kcol = kwbase + jb + jl;         // key seq pos
                v8f s;
#pragma unroll
                for (int r = 0; r < 8; ++r) s[r] = 0.0f;
#pragma unroll
                for (int ks = 0; ks < 4; ++ks) {
                    F16Frag kB;                            // contiguous 32B LDS read
                    kB.u[0] = *(const uint4*)&k_lds[jl][ks * 32 + hi * 16];
                    kB.u[1] = *(const uint4*)&k_lds[jl][ks * 32 + hi * 16 + 8];
                    s = wmma_f16(qA[ks].v, kB.v, s);
                }
                const unsigned char mk = mask[b * SS + kcol];
#pragma unroll
                for (int r = 0; r < 8; ++r) {
                    const int qs = q0 + r + 8 * hi;
                    float val = s[r] + attn_bias[((long long)b * SS + qs) * SS + kcol];
                    if (!mk) val = -3.0e38f;
                    s[r] = val;
                    tmax[r] = fmaxf(tmax[r], val);
                }
                sfr[js] = s;
            }

            // ---- row-max across the 16 lanes of each half-wave ----
#pragma unroll
            for (int r = 0; r < 8; ++r)
#pragma unroll
                for (int off = 1; off < 16; off <<= 1)
                    tmax[r] = fmaxf(tmax[r], __shfl_xor(tmax[r], off, 16));

            float fac[8];
#pragma unroll
            for (int r = 0; r < 8; ++r) {
                const float mnew = fmaxf(m_run[r], tmax[r]);
                fac[r]   = __expf(m_run[r] - mnew);
                m_run[r] = mnew;
            }

            // ---- exponentiate, write P tile to LDS, accumulate row sums ----
            float rsum[8];
#pragma unroll
            for (int r = 0; r < 8; ++r) rsum[r] = 0.0f;
#pragma unroll
            for (int js = 0; js < 2; ++js) {
                v8f s = sfr[js];
#pragma unroll
                for (int r = 0; r < 8; ++r) {
                    const float p = __expf(s[r] - m_run[r]);
                    rsum[r] += p;
                    p_lds[wave][r + 8 * hi][js * 16 + c] = (_Float16)p;
                }
            }
#pragma unroll
            for (int r = 0; r < 8; ++r) {
#pragma unroll
                for (int off = 1; off < 16; off <<= 1)
                    rsum[r] += __shfl_xor(rsum[r], off, 16);
                l_run[r] = l_run[r] * fac[r] + rsum[r];
            }

            // ---- rescale output accumulators ----
#pragma unroll
            for (int dt = 0; dt < 8; ++dt)
#pragma unroll
                for (int r = 0; r < 8; ++r) acc[dt][r] *= fac[r];

            __syncthreads();   // uniform: P writes visible for re-layout load

            // ---- P as A-fragment (lane=row=c, K = local key index) ----
            F16Frag pA;
            pA.u[0] = *(const uint4*)&p_lds[wave][c][hi * 16];
            pA.u[1] = *(const uint4*)&p_lds[wave][c][hi * 16 + 8];

            // ---- acc += P @ V  (8 d-tiles; B-fragment = 32B read of Vt) ----
#pragma unroll
            for (int dt = 0; dt < 8; ++dt) {
                F16Frag vB;
                vB.u[0] = *(const uint4*)&vt_lds[dt * 16 + c][hi * 16];
                vB.u[1] = *(const uint4*)&vt_lds[dt * 16 + c][hi * 16 + 8];
                acc[dt] = wmma_f16(pA.v, vB.v, acc[dt]);
            }
        }
    }

    // ---- normalize, gate, store f16 ----
    float inv[8];
#pragma unroll
    for (int r = 0; r < 8; ++r) inv[r] = 1.0f / l_run[r];
#pragma unroll
    for (int dt = 0; dt < 8; ++dt) {
#pragma unroll
        for (int r = 0; r < 8; ++r) {
            const int qs   = q0 + r + 8 * hi;
            const int dcol = head * DH + dt * 16 + c;
            const long long idx = (browbase + qs) * DI_ + dcol;
            const float g = (float)g16[idx];
            attout[idx] = (_Float16)(acc[dt][r] * inv[r] * g);
        }
    }
}

// ---------------------------------------------------------------------------
// Workspace layout (bytes, all 256B aligned). Total ~83.3 MB.
// ---------------------------------------------------------------------------
#define OFF_SEQ16  0LL                                    // 8192*128  f16 = 2 MB
#define OFF_WQ16   (OFF_SEQ16 + 2097152LL)                // 1024*128  f16
#define OFF_WKV16  (OFF_WQ16  + 262144LL)                 // 2048*128  f16
#define OFF_WG16   (OFF_WKV16 + 524288LL)                 // 1024*128  f16
#define OFF_WO16   (OFF_WG16  + 262144LL)                 // 128*1024  f16
#define OFF_Q16    (OFF_WO16  + 262144LL)                 // 8192*1024 f16 = 16 MB
#define OFF_K16    (OFF_Q16   + 16777216LL)
#define OFF_V16    (OFF_K16   + 16777216LL)
#define OFF_G16    (OFF_V16   + 16777216LL)
#define OFF_AO16   (OFF_G16   + 16777216LL)
#define WS_NEEDED  (OFF_AO16  + 16777216LL)

extern "C" void kernel_launch(void* const* d_in, const int* in_sizes, int n_in,
                              void* d_out, int out_size, void* d_ws, size_t ws_size,
                              hipStream_t stream) {
    (void)in_sizes; (void)n_in; (void)out_size; (void)ws_size;

    const float*         seq  = (const float*)d_in[0];
    const unsigned char* mask = (const unsigned char*)d_in[1];  // jnp.bool_ = 1 byte
    const float*         bias = (const float*)d_in[2];          // attn_bias [B,S,S]
    const float*         Wq   = (const float*)d_in[3];
    const float*         bq   = (const float*)d_in[4];
    const float*         Wkv  = (const float*)d_in[5];
    const float*         Wg   = (const float*)d_in[6];
    const float*         bg   = (const float*)d_in[7];
    const float*         Wo   = (const float*)d_in[8];
    float*               out  = (float*)d_out;

    char* ws = (char*)d_ws;
    _Float16* seq16 = (_Float16*)(ws + OFF_SEQ16);
    _Float16* Wq16  = (_Float16*)(ws + OFF_WQ16);
    _Float16* Wkv16 = (_Float16*)(ws + OFF_WKV16);
    _Float16* Wg16  = (_Float16*)(ws + OFF_WG16);
    _Float16* Wo16  = (_Float16*)(ws + OFF_WO16);
    _Float16* q16   = (_Float16*)(ws + OFF_Q16);
    _Float16* k16   = (_Float16*)(ws + OFF_K16);
    _Float16* v16   = (_Float16*)(ws + OFF_V16);
    _Float16* g16   = (_Float16*)(ws + OFF_G16);
    _Float16* ao16  = (_Float16*)(ws + OFF_AO16);

    // 1) f32 -> f16 conversions
    cvt_f32_to_f16<<<2048, 256, 0, stream>>>(seq, seq16, (long long)MROWS * DD);
    cvt_f32_to_f16<<<512,  256, 0, stream>>>(Wq,  Wq16,  (long long)DI_ * DD);
    cvt_f32_to_f16<<<1024, 256, 0, stream>>>(Wkv, Wkv16, (long long)2 * DI_ * DD);
    cvt_f32_to_f16<<<512,  256, 0, stream>>>(Wg,  Wg16,  (long long)DI_ * DD);
    cvt_f32_to_f16<<<512,  256, 0, stream>>>(Wo,  Wo16,  (long long)DD * DI_);

    // 2) projections: [8192,128] @ [128,1024]^T  -> 8192 wave-tiles -> 1024 blocks
    const float qscale = 0.08838834764831845f;   // 1/sqrt(128)
    gemm_wmma_kernel<<<1024, 256, 0, stream>>>(seq16, Wq16, bq, q16,
                                               MROWS, DI_, DD, qscale, 1);
    gemm_wmma_kernel<<<1024, 256, 0, stream>>>(seq16, Wkv16, nullptr, k16,
                                               MROWS, DI_, DD, 1.0f, 0);
    gemm_wmma_kernel<<<1024, 256, 0, stream>>>(seq16, Wkv16 + (long long)DI_ * DD, nullptr, v16,
                                               MROWS, DI_, DD, 1.0f, 0);
    gemm_wmma_kernel<<<1024, 256, 0, stream>>>(seq16, Wg16, bg, g16,
                                               MROWS, DI_, DD, 1.0f, 2);

    // 3) windowed attention: B*NH*NW = 512 blocks of 256 threads
    attn_window_kernel<<<BB * NH * NW, 256, 0, stream>>>(q16, k16, v16, g16,
                                                         bias, mask, ao16);

    // 4) output projection: [8192,1024] @ [1024,128]^T -> 1024 wave-tiles -> 128 blocks
    gemm_wmma_kernel<<<128, 256, 0, stream>>>(ao16, Wo16, nullptr, out,
                                              MROWS, DD, DI_, 1.0f, 3);
}
